// GemmaAttention_29721173688941
// MI455X (gfx1250) — matrix operand order repaired and checked by reading, not compiled, not run on hardware
//
#include <hip/hip_runtime.h>

// ---- problem constants (match reference) ----
#define B_    2
#define S_    2048
#define HID_  2048
#define H_    8
#define HD_   256
#define NH2_  128          // HD/2
#define SCALE_ 0.0625f     // HD^-0.5
#define PAD_  40           // LDS row pitch in elements (80B bf16 rows: 16B-aligned, conflict-free)

typedef __bf16 bf16;
typedef __attribute__((ext_vector_type(16))) __bf16 v16bf;
typedef __attribute__((ext_vector_type(8)))  __bf16 v8bf;
typedef __attribute__((ext_vector_type(8)))  float  v8f;
typedef __attribute__((ext_vector_type(4)))  float  v4f;

__device__ __forceinline__ bf16 f2bf(float f) {
  unsigned u = __builtin_bit_cast(unsigned, f);
  unsigned r = u + 0x7FFFu + ((u >> 16) & 1u);   // round-to-nearest-even
  unsigned short h = (unsigned short)(r >> 16);
  return __builtin_bit_cast(bf16, h);
}

// CDNA5 async global->LDS copy (ASYNCcnt-tracked); 16B per active lane.
// VDST = per-lane LDS byte address, VADDR = per-lane 64-bit global address.
__device__ __forceinline__ void async_b128(unsigned lds_addr, const void* gaddr) {
  asm volatile("global_load_async_to_lds_b128 %0, %1, off"
               :: "v"(lds_addr), "v"(gaddr) : "memory");
}
__device__ __forceinline__ void wait_async0() {
  asm volatile("s_wait_asynccnt 0" ::: "memory");
}
__device__ __forceinline__ unsigned lds_addr_of(const void* p) {
  // generic LDS addr: addr[31:0] is the LDS byte offset (ISA 10.2 aperture rules)
  return (unsigned)(size_t)p;
}

__device__ __forceinline__ v8f wmma_bf16(v16bf a, v16bf b, v8f c) {
  return __builtin_amdgcn_wmma_f32_16x16x32_bf16(false, a, false, b, (short)0, c, false, false);
}

__device__ __forceinline__ v16bf ld_frag16(const bf16* p) {
  v8bf x0 = *(const v8bf*)(p);
  v8bf x1 = *(const v8bf*)(p + 8);
  v16bf v;
#pragma unroll
  for (int i = 0; i < 8; ++i) { v[i] = x0[i]; v[8 + i] = x1[i]; }
  return v;
}

// ---------------------------------------------------------------------------
// Async-LDS double-buffered bf16 WMMA GEMM:
//   C[M,N] (f32) = A[M,K] (bf16 row-major) x Bt[N,K]^T (bf16 row-major)
// Block: 256 threads (8 waves), tile 128x64, K-step 32.
// Wave w: rows [w*16, w*16+16), all 64 cols -> 4 WMMA tiles, 1 shared A frag.
// All four B fragments are loaded into distinct registers BEFORE the WMMA
// chain so ds_load latency overlaps the matrix pipe.
// M % 128 == 0, N % 64 == 0, K % 32 == 0 (all true for this problem).
// ---------------------------------------------------------------------------
__global__ void __launch_bounds__(256) gemm_bf16_f32(
    const bf16* __restrict__ A, const bf16* __restrict__ Bt, float* __restrict__ C,
    int M, int N, int K, int lda, int ldb, int ldc)
{
  __shared__ bf16 sA[2][128 * PAD_];
  __shared__ bf16 sB[2][64 * PAD_];

  const int tid  = threadIdx.x;
  const int lane = tid & 31;
  const int wave = tid >> 5;
  const int l16  = lane & 15;
  const int half = lane >> 4;
  const int mBase = blockIdx.y * 128;
  const int nBase = blockIdx.x * 64;

  // staging map: A tile 128x32 = 512 x 16B chunks (2/thread); B tile 64x32 = 256 chunks
  const int ar0 = tid >> 2,         ac0 = (tid & 3) * 8;
  const int ar1 = (tid + 256) >> 2, ac1 = ((tid + 256) & 3) * 8;
  const int br  = tid >> 2,         bc  = (tid & 3) * 8;

  const bf16* gA0 = A  + (size_t)(mBase + ar0) * lda + ac0;
  const bf16* gA1 = A  + (size_t)(mBase + ar1) * lda + ac1;
  const bf16* gB  = Bt + (size_t)(nBase + br)  * ldb + bc;

  const unsigned lA0_0 = lds_addr_of(&sA[0][ar0 * PAD_ + ac0]);
  const unsigned lA0_1 = lds_addr_of(&sA[1][ar0 * PAD_ + ac0]);
  const unsigned lA1_0 = lds_addr_of(&sA[0][ar1 * PAD_ + ac1]);
  const unsigned lA1_1 = lds_addr_of(&sA[1][ar1 * PAD_ + ac1]);
  const unsigned lB_0  = lds_addr_of(&sB[0][br * PAD_ + bc]);
  const unsigned lB_1  = lds_addr_of(&sB[1][br * PAD_ + bc]);

  v8f acc0 = {}, acc1 = {}, acc2 = {}, acc3 = {};

  // prefetch k0 = 0 into buffer 0
  async_b128(lA0_0, gA0);
  async_b128(lA1_0, gA1);
  async_b128(lB_0,  gB);

  int buf = 0;
  const int aoff = (wave * 16 + l16) * PAD_ + half * 16;
  const int boff = l16 * PAD_ + half * 16;
  for (int k0 = 0; k0 < K; k0 += 32) {
    wait_async0();
    __syncthreads();
    if (k0 + 32 < K) {
      const int kn = k0 + 32;
      async_b128(buf ? lA0_0 : lA0_1, gA0 + kn);
      async_b128(buf ? lA1_0 : lA1_1, gA1 + kn);
      async_b128(buf ? lB_0  : lB_1,  gB  + kn);
    }
    const bf16* sAb = buf ? &sA[1][0] : &sA[0][0];
    const bf16* sBb = buf ? &sB[1][0] : &sB[0][0];

    // issue all fragment loads first, then drain into the matrix pipe
    v16bf va  = ld_frag16(sAb + aoff);          // elems 0..15: K = half*16 + s
    v16bf vb0 = ld_frag16(sBb + boff);
    v16bf vb1 = ld_frag16(sBb + boff + 16 * PAD_);
    v16bf vb2 = ld_frag16(sBb + boff + 32 * PAD_);
    v16bf vb3 = ld_frag16(sBb + boff + 48 * PAD_);

    acc0 = wmma_bf16(va, vb0, acc0);
    acc1 = wmma_bf16(va, vb1, acc1);
    acc2 = wmma_bf16(va, vb2, acc2);
    acc3 = wmma_bf16(va, vb3, acc3);

    __syncthreads();
    buf ^= 1;
  }

  const int m0 = mBase + wave * 16;
#pragma unroll
  for (int j = 0; j < 8; ++j) {
    float* cr = C + (size_t)(m0 + half * 8 + j) * ldc + nBase + l16;
    cr[0]  = acc0[j];
    cr[16] = acc1[j];
    cr[32] = acc2[j];
    cr[48] = acc3[j];
  }
}

// NOTE on A-fragment layout: ld_frag16(sAb + aoff) yields elems 0..15 = K
// columns half*16..half*16+15 (contiguous K, the same order the B fragment uses;
// not the ISA 7.12.2 table order).  B fragment rows are N-columns of
// C (Bt rows), elems 0..15 = 16 contiguous K starting at half*16.

// ---------------------------------------------------------------------------
// scores[bh,s,t] = (Q_bh[s,:] . K_b[t,:]) * scale + mask[b,s,t]
// Same async-LDS structure; K = HD = 256 (8 steps).
// ---------------------------------------------------------------------------
__global__ void __launch_bounds__(256) attn_scores_kernel(
    const bf16* __restrict__ Qrot, const bf16* __restrict__ Krot,
    const float* __restrict__ mask, float* __restrict__ attn)
{
  __shared__ bf16 sA[2][128 * PAD_];
  __shared__ bf16 sB[2][64 * PAD_];

  const int bh = blockIdx.z;
  const int b  = bh / H_;
  const bf16*  A  = Qrot + (size_t)bh * S_ * HD_;
  const bf16*  Bt = Krot + (size_t)b  * S_ * HD_;
  const float* Mk = mask + (size_t)b  * S_ * S_;
  float*       Co = attn + (size_t)bh * S_ * S_;

  const int tid  = threadIdx.x;
  const int lane = tid & 31;
  const int wave = tid >> 5;
  const int l16  = lane & 15;
  const int half = lane >> 4;
  const int mBase = blockIdx.y * 128;
  const int nBase = blockIdx.x * 64;

  const int ar0 = tid >> 2,         ac0 = (tid & 3) * 8;
  const int ar1 = (tid + 256) >> 2, ac1 = ((tid + 256) & 3) * 8;
  const int br  = tid >> 2,         bc  = (tid & 3) * 8;

  const bf16* gA0 = A  + (size_t)(mBase + ar0) * HD_ + ac0;
  const bf16* gA1 = A  + (size_t)(mBase + ar1) * HD_ + ac1;
  const bf16* gB  = Bt + (size_t)(nBase + br)  * HD_ + bc;

  const unsigned lA0_0 = lds_addr_of(&sA[0][ar0 * PAD_ + ac0]);
  const unsigned lA0_1 = lds_addr_of(&sA[1][ar0 * PAD_ + ac0]);
  const unsigned lA1_0 = lds_addr_of(&sA[0][ar1 * PAD_ + ac1]);
  const unsigned lA1_1 = lds_addr_of(&sA[1][ar1 * PAD_ + ac1]);
  const unsigned lB_0  = lds_addr_of(&sB[0][br * PAD_ + bc]);
  const unsigned lB_1  = lds_addr_of(&sB[1][br * PAD_ + bc]);

  v8f acc0 = {}, acc1 = {}, acc2 = {}, acc3 = {};

  async_b128(lA0_0, gA0);
  async_b128(lA1_0, gA1);
  async_b128(lB_0,  gB);

  int buf = 0;
  const int aoff = (wave * 16 + l16) * PAD_ + half * 16;
  const int boff = l16 * PAD_ + half * 16;
  for (int k0 = 0; k0 < HD_; k0 += 32) {
    wait_async0();
    __syncthreads();
    if (k0 + 32 < HD_) {
      const int kn = k0 + 32;
      async_b128(buf ? lA0_0 : lA0_1, gA0 + kn);
      async_b128(buf ? lA1_0 : lA1_1, gA1 + kn);
      async_b128(buf ? lB_0  : lB_1,  gB  + kn);
    }
    const bf16* sAb = buf ? &sA[1][0] : &sA[0][0];
    const bf16* sBb = buf ? &sB[1][0] : &sB[0][0];

    v16bf va  = ld_frag16(sAb + aoff);
    v16bf vb0 = ld_frag16(sBb + boff);
    v16bf vb1 = ld_frag16(sBb + boff + 16 * PAD_);
    v16bf vb2 = ld_frag16(sBb + boff + 32 * PAD_);
    v16bf vb3 = ld_frag16(sBb + boff + 48 * PAD_);

    acc0 = wmma_bf16(va, vb0, acc0);
    acc1 = wmma_bf16(va, vb1, acc1);
    acc2 = wmma_bf16(va, vb2, acc2);
    acc3 = wmma_bf16(va, vb3, acc3);

    __syncthreads();
    buf ^= 1;
  }

  const int m0 = mBase + wave * 16;
#pragma unroll
  for (int j = 0; j < 8; ++j) {
    const size_t r = (size_t)(m0 + half * 8 + j) * S_;
    const int c0 = nBase + l16;
    Co[r + c0]      = acc0[j] * SCALE_ + Mk[r + c0];
    Co[r + c0 + 16] = acc1[j] * SCALE_ + Mk[r + c0 + 16];
    Co[r + c0 + 32] = acc2[j] * SCALE_ + Mk[r + c0 + 32];
    Co[r + c0 + 48] = acc3[j] * SCALE_ + Mk[r + c0 + 48];
  }
}

// ---------------------------------------------------------------------------
// Row softmax in place over attn rows of length S (B*H*S rows), vectorized.
// ---------------------------------------------------------------------------
__global__ void __launch_bounds__(256) softmax_inplace(float* __restrict__ attn)
{
  __shared__ float red[256];
  float* p = attn + (size_t)blockIdx.x * S_;
  const int tid = threadIdx.x;

  float mx = -3.4e38f;
  for (int i = tid * 4; i < S_; i += 1024) {
    v4f x = *(const v4f*)(p + i);
    mx = fmaxf(mx, fmaxf(fmaxf(x[0], x[1]), fmaxf(x[2], x[3])));
  }
  red[tid] = mx; __syncthreads();
  for (int off = 128; off > 0; off >>= 1) {
    if (tid < off) red[tid] = fmaxf(red[tid], red[tid + off]);
    __syncthreads();
  }
  mx = red[0]; __syncthreads();

  float sum = 0.f;
  for (int i = tid * 4; i < S_; i += 1024) {
    v4f x = *(const v4f*)(p + i);
#pragma unroll
    for (int j = 0; j < 4; ++j) { x[j] = __expf(x[j] - mx); sum += x[j]; }
    *(v4f*)(p + i) = x;
  }
  red[tid] = sum; __syncthreads();
  for (int off = 128; off > 0; off >>= 1) {
    if (tid < off) red[tid] += red[tid + off];
    __syncthreads();
  }
  const float inv = 1.0f / red[0];
  for (int i = tid * 4; i < S_; i += 1024) {
    v4f x = *(const v4f*)(p + i);
#pragma unroll
    for (int j = 0; j < 4; ++j) x[j] *= inv;
    *(v4f*)(p + i) = x;
  }
}

// ---------------------------------------------------------------------------
// out_heads[b,s,h*HD+d] = sum_t attn[bh,s,t] * V[b,t,d]
// Block tile 32 x 256 (full head dim in N) so the 268MB attn tensor streams
// from memory exactly once; attn f32 staged via async-LDS, converted to bf16
// fragments in registers. Vt (tiny, L2-resident) read directly.
// Wave w: rows (w&1)*16, cols (w>>1)*64 (4 WMMA tiles).
// ---------------------------------------------------------------------------
#define APF_ 40  // f32 LDS pitch
__global__ void __launch_bounds__(256) attn_av_kernel(
    const float* __restrict__ attn, const bf16* __restrict__ Vt,
    bf16* __restrict__ outh)
{
  __shared__ float sAf[2][32 * APF_];

  const int bh = blockIdx.z;
  const int b = bh / H_, h = bh % H_;
  const float* A  = attn + (size_t)bh * S_ * S_;
  const bf16*  Bt = Vt   + (size_t)b * HD_ * S_;

  const int tid  = threadIdx.x;
  const int lane = tid & 31;
  const int wave = tid >> 5;
  const int l16  = lane & 15;
  const int half = lane >> 4;
  const int mBase = blockIdx.y * 32;
  const int mw = wave & 1;
  const int nw = wave >> 1;           // 0..3 -> n base nw*64

  // staging map: A tile 32 rows x 32 f32 = 256 x 16B chunks (1/thread)
  const int ar = tid >> 3, ac = (tid & 7) * 4;
  const float* gA = A + (size_t)(mBase + ar) * S_ + ac;
  const unsigned lA_0 = lds_addr_of(&sAf[0][ar * APF_ + ac]);
  const unsigned lA_1 = lds_addr_of(&sAf[1][ar * APF_ + ac]);

  v8f acc0 = {}, acc1 = {}, acc2 = {}, acc3 = {};

  async_b128(lA_0, gA);

  int buf = 0;
  const int aoff = (mw * 16 + l16) * APF_ + half * 16;
  const bf16* gBrow = Bt + (size_t)(nw * 64 + l16) * S_ + half * 16;
  for (int k0 = 0; k0 < S_; k0 += 32) {
    wait_async0();
    __syncthreads();
    if (k0 + 32 < S_) async_b128(buf ? lA_0 : lA_1, gA + k0 + 32);

    // issue the 4 global B loads first (L2-resident), then LDS A reads
    v16bf vb0 = *(const v16bf*)(gBrow + k0);
    v16bf vb1 = *(const v16bf*)(gBrow + (size_t)16 * S_ + k0);
    v16bf vb2 = *(const v16bf*)(gBrow + (size_t)32 * S_ + k0);
    v16bf vb3 = *(const v16bf*)(gBrow + (size_t)48 * S_ + k0);

    const float* sa = (buf ? &sAf[1][0] : &sAf[0][0]) + aoff;
    v4f f0 = *(const v4f*)(sa);
    v4f f1 = *(const v4f*)(sa + 4);
    v4f f2 = *(const v4f*)(sa + 8);
    v4f f3 = *(const v4f*)(sa + 12);
    v16bf va;
#pragma unroll
    for (int i = 0; i < 4; ++i) {
      va[i]      = f2bf(f0[i]);  va[4 + i]  = f2bf(f1[i]);
      va[8 + i]  = f2bf(f2[i]);  va[12 + i] = f2bf(f3[i]);
    }

    acc0 = wmma_bf16(va, vb0, acc0);
    acc1 = wmma_bf16(va, vb1, acc1);
    acc2 = wmma_bf16(va, vb2, acc2);
    acc3 = wmma_bf16(va, vb3, acc3);

    __syncthreads();
    buf ^= 1;
  }

#pragma unroll
  for (int j = 0; j < 8; ++j) {
    const int row = mBase + mw * 16 + half * 8 + j;      // s
    bf16* o = outh + ((size_t)b * S_ + row) * HID_ + h * HD_ + nw * 64 + l16;
    o[0]  = f2bf(acc0[j]);
    o[16] = f2bf(acc1[j]);
    o[32] = f2bf(acc2[j]);
    o[48] = f2bf(acc3[j]);
  }
}

// ---------------------------------------------------------------------------
// RoPE on Q/K (f32 in) -> bf16 packed layouts; V f32 -> bf16 [B, HD, S].
// ---------------------------------------------------------------------------
__global__ void __launch_bounds__(256) rope_pack(
    const float* __restrict__ Qf, const float* __restrict__ Kf,
    const float* __restrict__ Vf, const int* __restrict__ pos_ids,
    bf16* __restrict__ Qrot, bf16* __restrict__ Krot, bf16* __restrict__ Vt)
{
  const int bs = blockIdx.x;            // b*S + s
  const int b = bs / S_, s = bs % S_;
  const float pos = (float)pos_ids[bs];

  for (int idx = threadIdx.x; idx < H_ * NH2_; idx += 256) {
    const int h = idx / NH2_, d = idx % NH2_;
    const float inv = __powf(10000.0f, -((float)d) / (float)NH2_);
    const float fr = pos * inv;
    const float c = cosf(fr), sn = sinf(fr);
    const float* q = Qf + (size_t)bs * HID_ + h * HD_;
    const float q1 = q[d], q2 = q[d + NH2_];
    bf16* o = Qrot + ((size_t)(b * H_ + h) * S_ + s) * HD_;
    o[d]        = f2bf(q1 * c - q2 * sn);
    o[d + NH2_] = f2bf(q2 * c + q1 * sn);
  }
  for (int d = threadIdx.x; d < NH2_; d += 256) {
    const float inv = __powf(10000.0f, -((float)d) / (float)NH2_);
    const float fr = pos * inv;
    const float c = cosf(fr), sn = sinf(fr);
    const float* kk = Kf + (size_t)bs * HD_;
    const float k1 = kk[d], k2 = kk[d + NH2_];
    bf16* o = Krot + (size_t)bs * HD_;
    o[d]        = f2bf(k1 * c - k2 * sn);
    o[d + NH2_] = f2bf(k2 * c + k1 * sn);
  }
  for (int d = threadIdx.x; d < HD_; d += 256) {
    Vt[((size_t)b * HD_ + d) * S_ + s] = f2bf(Vf[(size_t)bs * HD_ + d]);
  }
}

// ---- one-time conversion kernels ----
__global__ void __launch_bounds__(256) cast_bf16(
    const float* __restrict__ src, bf16* __restrict__ dst, int n)
{
  int i = blockIdx.x * 256 + threadIdx.x;
  const int stride = gridDim.x * 256;
  for (; i < n; i += stride) dst[i] = f2bf(src[i]);
}

// src: K x N row-major f32 -> dst: N x K row-major bf16 (i.e. B^T)
__global__ void __launch_bounds__(256) transpose_cast(
    const float* __restrict__ src, bf16* __restrict__ dst, int K, int N)
{
  size_t t = (size_t)blockIdx.x * 256 + threadIdx.x;
  const size_t total = (size_t)K * N;
  const size_t stride = (size_t)gridDim.x * 256;
  for (; t < total; t += stride) {
    const size_t n = t / (size_t)K, k = t % (size_t)K;
    dst[t] = f2bf(src[k * (size_t)N + n]);
  }
}

// ---------------------------------------------------------------------------
extern "C" void kernel_launch(void* const* d_in, const int* in_sizes, int n_in,
                              void* d_out, int out_size, void* d_ws, size_t ws_size,
                              hipStream_t stream)
{
  (void)in_sizes; (void)n_in; (void)out_size; (void)ws_size;
  const float* hs   = (const float*)d_in[0];
  const float* mask = (const float*)d_in[1];
  const int*   pos  = (const int*)d_in[2];
  const float* Wq   = (const float*)d_in[3];
  const float* Wk   = (const float*)d_in[4];
  const float* Wv   = (const float*)d_in[5];
  const float* Wo   = (const float*)d_in[6];

  float* out  = (float*)d_out;                          // [B,S,HID]
  float* attn = (float*)d_out + (size_t)B_ * S_ * HID_; // [B,H,S,S]

  // workspace carve-up (~115 MB, 256B-aligned slices)
  char* w = (char*)d_ws;
  auto alloc = [&](size_t bytes) -> char* {
    char* p = w; w += (bytes + 255) & ~(size_t)255; return p;
  };
  bf16*  hs_bf = (bf16*)alloc((size_t)B_ * S_ * HID_ * 2);
  bf16*  Wq_t  = (bf16*)alloc((size_t)HID_ * HID_ * 2);
  bf16*  Wk_t  = (bf16*)alloc((size_t)HD_ * HID_ * 2);
  bf16*  Wv_t  = (bf16*)alloc((size_t)HD_ * HID_ * 2);
  bf16*  Wo_t  = (bf16*)alloc((size_t)HID_ * HID_ * 2);
  float* Qf    = (float*)alloc((size_t)B_ * S_ * HID_ * 4);
  float* Kf    = (float*)alloc((size_t)B_ * S_ * HD_ * 4);
  float* Vf    = (float*)alloc((size_t)B_ * S_ * HD_ * 4);
  bf16*  Qrot  = (bf16*)alloc((size_t)B_ * H_ * S_ * HD_ * 2);
  bf16*  Krot  = (bf16*)alloc((size_t)B_ * S_ * HD_ * 2);
  bf16*  Vt    = (bf16*)alloc((size_t)B_ * HD_ * S_ * 2);
  bf16*  outh  = (bf16*)alloc((size_t)B_ * S_ * HID_ * 2);

  const int Mrows = B_ * S_;   // 4096
  const dim3 blk(256);

  // precision conversion + weight transposes (one-time streaming pass)
  cast_bf16<<<4096, blk, 0, stream>>>(hs, hs_bf, B_ * S_ * HID_);
  transpose_cast<<<8192, blk, 0, stream>>>(Wq, Wq_t, HID_, HID_);
  transpose_cast<<<2048, blk, 0, stream>>>(Wk, Wk_t, HID_, HD_);
  transpose_cast<<<2048, blk, 0, stream>>>(Wv, Wv_t, HID_, HD_);
  transpose_cast<<<8192, blk, 0, stream>>>(Wo, Wo_t, HID_, HID_);

  // Q/K/V projections (async-LDS bf16 WMMA, f32 accumulate)
  gemm_bf16_f32<<<dim3(HID_ / 64, Mrows / 128, 1), blk, 0, stream>>>(
      hs_bf, Wq_t, Qf, Mrows, HID_, HID_, HID_, HID_, HID_);
  gemm_bf16_f32<<<dim3(HD_ / 64, Mrows / 128, 1), blk, 0, stream>>>(
      hs_bf, Wk_t, Kf, Mrows, HD_, HID_, HID_, HID_, HD_);
  gemm_bf16_f32<<<dim3(HD_ / 64, Mrows / 128, 1), blk, 0, stream>>>(
      hs_bf, Wv_t, Vf, Mrows, HD_, HID_, HID_, HID_, HD_);

  // RoPE + pack to attention-friendly bf16 layouts
  rope_pack<<<Mrows, blk, 0, stream>>>(Qf, Kf, Vf, pos, Qrot, Krot, Vt);

  // scores -> attn region of d_out, softmax in place, then AV
  attn_scores_kernel<<<dim3(S_ / 64, S_ / 128, B_ * H_), blk, 0, stream>>>(
      Qrot, Krot, mask, attn);
  softmax_inplace<<<B_ * H_ * S_, blk, 0, stream>>>(attn);
  attn_av_kernel<<<dim3(1, S_ / 32, B_ * H_), blk, 0, stream>>>(
      attn, Vt, outh);

  // output projection -> out region of d_out
  gemm_bf16_f32<<<dim3(HID_ / 64, Mrows / 128, 1), blk, 0, stream>>>(
      outh, Wo_t, out, Mrows, HID_, HID_, HID_, HID_, HID_);
}
